// TaylorAttention_28991029248159
// MI455X (gfx1250) — matrix operand order repaired and checked by reading, not compiled, not run on hardware
//
#include <hip/hip_runtime.h>

#define NSEQ 577
#define NPAD 608            // multiple of 32 (19 chunks of 32)
#define BATCH 16
#define NHEAD 12
#define HDIM 64
#define CDIM 768
#define NTOK (BATCH * NSEQ) // 9232 = 577 * 16 (exact multiple of 16)
#define BH (BATCH * NHEAD)  // 192

typedef __attribute__((ext_vector_type(16))) _Float16 v16h;
typedef __attribute__((ext_vector_type(8)))  _Float16 v8h;
typedef __attribute__((ext_vector_type(8)))  float    v8f;
typedef __attribute__((ext_vector_type(4)))  float    v4f;
typedef __attribute__((ext_vector_type(4)))  _Float16 v4h;

union V16H { v16h v; v8h h[2]; };

#if defined(__has_builtin) && __has_builtin(__builtin_amdgcn_sched_barrier)
#define SCHED_FENCE() __builtin_amdgcn_sched_barrier(0)
#else
#define SCHED_FENCE()
#endif

// A-matrix fragment, 16x32 f16 (ISA 05_wmma "16-bit A-Matrix 16x32"):
// lane L: row m = L&15; elements 0..7 = K 8g..8g+7, elements 8..15 = K 16+8g..16+8g+7 (g = L>>4).
__device__ __forceinline__ v16h load_fragA(const _Float16* tile, int ldh) {
  const int lane = threadIdx.x & 31;
  const int m = lane & 15;
  const int g = lane >> 4;
  const _Float16* p = tile + m * ldh + g * 8;
  V16H u;
  u.h[0] = *(const v8h*)(p);
  u.h[1] = *(const v8h*)(p + 16);
  return u.v;
}

// B-matrix fragment, 32x16 f16, loaded from the *transposed* tile BT[n][k] (row-major, ld = ldh):
// ISA: B rows striped across lanes within one VGPR -> lane L: col n = L&15,
// elements 0..7 = K 8g..8g+7, elements 8..15 = K 16+8g..16+8g+7 (same order as A).
__device__ __forceinline__ v16h load_fragB(const _Float16* tileT, int ldh) {
  const int lane = threadIdx.x & 31;
  const int n = lane & 15;
  const int g = lane >> 4;
  const _Float16* p = tileT + n * ldh + g * 8;
  V16H u;
  u.h[0] = *(const v8h*)(p);
  u.h[1] = *(const v8h*)(p + 16);
  return u.v;
}

__device__ __forceinline__ v8f wmma16(v16h a, v16h b, v8f c) {
  return __builtin_amdgcn_wmma_f32_16x16x32_f16(false, a, false, b, (short)0, c, false, false);
}

__constant__ const v8f kZero8 = {0.f,0.f,0.f,0.f,0.f,0.f,0.f,0.f};

// ---------------- f32 -> f16 convert (x4 vectorized; counts are multiples of 4) ----------------
__global__ void k_cvt(const float* __restrict__ src, _Float16* __restrict__ dst, int n4) {
  int i = blockIdx.x * blockDim.x + threadIdx.x;
  int stride = gridDim.x * blockDim.x;
  for (; i < n4; i += stride) {
    v4f f = ((const v4f*)src)[i];
    v4h h;
    h.x = (_Float16)f.x; h.y = (_Float16)f.y; h.z = (_Float16)f.z; h.w = (_Float16)f.w;
    ((v4h*)dst)[i] = h;
  }
}

// ---------------- QKV GEMM: [9232,768] x [2304,768]^T, route to q (scaled), k, v^T ----------------
// block = 128 threads (4 waves); wave computes 16 tokens x 64 out-cols (4 WMMA tiles, A reused),
// k-loop software-pipelined: next step's fragments prefetched before this step's 4 WMMAs.
__global__ void k_qkv(const _Float16* __restrict__ xh,
                      const _Float16* __restrict__ wh,
                      _Float16* __restrict__ qh,   // [BH, NPAD, 64]
                      _Float16* __restrict__ kh,   // [BH, NPAD, 64]
                      _Float16* __restrict__ vth)  // [BH, 64, NPAD]
{
  const int wave = threadIdx.x >> 5;
  const int lane = threadIdx.x & 31;
  const int mTile = blockIdx.x % (NTOK / 16);
  const int nBlk  = blockIdx.x / (NTOK / 16);
  const int m0 = mTile * 16;
  const int n0 = nBlk * 256 + wave * 64;

  const _Float16* aBase = xh + (size_t)m0 * CDIM;
  const _Float16* bBase = wh + (size_t)n0 * CDIM;

  v8f acc[4] = {kZero8, kZero8, kZero8, kZero8};
  v16h a  = load_fragA(aBase, CDIM);
  v16h b0 = load_fragB(bBase +  0 * CDIM, CDIM);
  v16h b1 = load_fragB(bBase + 16 * CDIM, CDIM);
  v16h b2 = load_fragB(bBase + 32 * CDIM, CDIM);
  v16h b3 = load_fragB(bBase + 48 * CDIM, CDIM);
  for (int k0 = 0; k0 < CDIM; k0 += 32) {
    const int kn = (k0 + 32 == CDIM) ? 0 : (k0 + 32);   // wrap: last prefetch harmless
    v16h an  = load_fragA(aBase + kn, CDIM);
    v16h bn0 = load_fragB(bBase + kn +  0 * CDIM, CDIM);
    v16h bn1 = load_fragB(bBase + kn + 16 * CDIM, CDIM);
    v16h bn2 = load_fragB(bBase + kn + 32 * CDIM, CDIM);
    v16h bn3 = load_fragB(bBase + kn + 48 * CDIM, CDIM);
    SCHED_FENCE();                                      // keep prefetch above the WMMAs
    acc[0] = wmma16(a, b0, acc[0]);
    acc[1] = wmma16(a, b1, acc[1]);
    acc[2] = wmma16(a, b2, acc[2]);
    acc[3] = wmma16(a, b3, acc[3]);
    a = an; b0 = bn0; b1 = bn1; b2 = bn2; b3 = bn3;
  }

  const int c = lane & 15, g = lane >> 4;
  const int seg  = n0 / CDIM;          // 0=q 1=k 2=v (wave's 64 cols sit in one segment+head)
  const int head = (n0 % CDIM) >> 6;
  #pragma unroll
  for (int r = 0; r < 8; ++r) {
    const int tok = m0 + r + 8 * g;
    const int bb  = tok / NSEQ;
    const int n   = tok - bb * NSEQ;
    const int bhI = bb * NHEAD + head;
    #pragma unroll
    for (int t = 0; t < 4; ++t) {
      const int d = t * 16 + c;
      const float val = acc[t][r];
      if (seg == 0)      qh[((size_t)bhI * NPAD + n) * HDIM + d] = (_Float16)(val * 0.125f);
      else if (seg == 1) kh[((size_t)bhI * NPAD + n) * HDIM + d] = (_Float16)val;
      else               vth[((size_t)bhI * HDIM + d) * NPAD + n] = (_Float16)val;
    }
  }
}

// ---------------- Fused Taylor attention (flash-style single pass) ----------------
// 1 wave per (b,h,16-query row tile); loop over 19 chunks of 32 keys:
//   S = Q(16x64) Kc^T (4 wmma) -> P = relu(2+S/2), masked; rowsum += P;
//   P staged via 1KB LDS -> A fragment; O += P Vc (4 wmma). Normalize + store at end.
// K-tile fragments for the next chunk are prefetched before this chunk's S WMMAs.
__global__ void __launch_bounds__(32) k_attn(const _Float16* __restrict__ qh,
                                             const _Float16* __restrict__ kh,
                                             const _Float16* __restrict__ vth,
                                             _Float16* __restrict__ aoh) // [B,N,768] = [B,N,H,D]
{
  __shared__ __align__(16) _Float16 pt[16 * 32];
  const int lane = threadIdx.x & 31;
  const int c = lane & 15, g = lane >> 4;
  const int nTiles = (NSEQ + 15) / 16;    // 37
  const int rowTile = blockIdx.x % nTiles;
  const int bh = blockIdx.x / nTiles;
  const int bb = bh / NHEAD, head = bh % NHEAD;

  const _Float16* qb = qh + ((size_t)bh * NPAD + rowTile * 16) * HDIM;
  const v16h a0 = load_fragA(qb +  0, HDIM);   // d 0..31
  const v16h a1 = load_fragA(qb + 32, HDIM);   // d 32..63

  const _Float16* kBase = kh  + (size_t)bh * NPAD * HDIM;
  const _Float16* vBase = vth + (size_t)bh * HDIM * NPAD;

  v8f oacc[4] = {kZero8, kZero8, kZero8, kZero8};
  float rs[8] = {0.f,0.f,0.f,0.f,0.f,0.f,0.f,0.f};

  v16h b00 = load_fragB(kBase, HDIM);                  // rows 0..15,  d 0..31
  v16h b01 = load_fragB(kBase + 32, HDIM);             // rows 0..15,  d 32..63
  v16h b10 = load_fragB(kBase + 16 * HDIM, HDIM);      // rows 16..31, d 0..31
  v16h b11 = load_fragB(kBase + 16 * HDIM + 32, HDIM); // rows 16..31, d 32..63

  for (int nc = 0; nc < NPAD; nc += 32) {
    const int ncn = (nc + 32 == NPAD) ? 0 : (nc + 32);  // wrap: last prefetch harmless
    const _Float16* kbn = kBase + (size_t)ncn * HDIM;
    v16h bn00 = load_fragB(kbn, HDIM);
    v16h bn01 = load_fragB(kbn + 32, HDIM);
    v16h bn10 = load_fragB(kbn + 16 * HDIM, HDIM);
    v16h bn11 = load_fragB(kbn + 16 * HDIM + 32, HDIM);
    SCHED_FENCE();

    v8f s0 = kZero8, s1 = kZero8;
    s0 = wmma16(a0, b00, s0);
    s0 = wmma16(a1, b01, s0);
    s1 = wmma16(a0, b10, s1);
    s1 = wmma16(a1, b11, s1);

    const float m0v = (nc + c)      < NSEQ ? 1.f : 0.f;  // mask padded key columns
    const float m1v = (nc + 16 + c) < NSEQ ? 1.f : 0.f;
    #pragma unroll
    for (int r = 0; r < 8; ++r) {
      float p0 = fmaxf(0.f, 2.f + 0.5f * s0[r]) * m0v;   // relu(2 + s/2)
      float p1 = fmaxf(0.f, 2.f + 0.5f * s1[r]) * m1v;
      rs[r] += p0 + p1;
      pt[(r + 8 * g) * 32 + c]      = (_Float16)p0;       // C-layout -> row-major 16x32
      pt[(r + 8 * g) * 32 + 16 + c] = (_Float16)p1;
    }
    __syncthreads();
    v16h pf = load_fragA(pt, 32);                         // P is the A operand of P@V
    #pragma unroll
    for (int t = 0; t < 4; ++t) {
      v16h bv = load_fragB(vBase + (size_t)(t * 16) * NPAD + nc, NPAD);
      oacc[t] = wmma16(pf, bv, oacc[t]);
    }
    __syncthreads();
    b00 = bn00; b01 = bn01; b10 = bn10; b11 = bn11;
  }

  // rowsum lives distributed over 16 lanes (one key-col each): reduce within half-wave
  #pragma unroll
  for (int r = 0; r < 8; ++r) {
    float v = rs[r];
    v += __shfl_xor(v, 1, 32);
    v += __shfl_xor(v, 2, 32);
    v += __shfl_xor(v, 4, 32);
    v += __shfl_xor(v, 8, 32);
    rs[r] = 1.f / v;
  }
  #pragma unroll
  for (int r = 0; r < 8; ++r) {
    const int n = rowTile * 16 + r + 8 * g;
    if (n < NSEQ) {
      #pragma unroll
      for (int t = 0; t < 4; ++t)
        aoh[((size_t)bb * NSEQ + n) * CDIM + head * HDIM + t * 16 + c] =
            (_Float16)(oacc[t][r] * rs[r]);
    }
  }
}

// ---------------- Output projection: [9232,768] x [768,768]^T + bias -> f32 ----------------
__global__ void k_proj(const _Float16* __restrict__ aoh,
                       const _Float16* __restrict__ wh,
                       const float* __restrict__ pb,
                       float* __restrict__ out)
{
  const int wave = threadIdx.x >> 5;
  const int lane = threadIdx.x & 31;
  const int mTile = blockIdx.x % (NTOK / 16);
  const int nBlk  = blockIdx.x / (NTOK / 16);
  const int m0 = mTile * 16;
  const int n0 = nBlk * 256 + wave * 64;

  const _Float16* aBase = aoh + (size_t)m0 * CDIM;
  const _Float16* bBase = wh + (size_t)n0 * CDIM;

  v8f acc[4] = {kZero8, kZero8, kZero8, kZero8};
  v16h a  = load_fragA(aBase, CDIM);
  v16h b0 = load_fragB(bBase +  0 * CDIM, CDIM);
  v16h b1 = load_fragB(bBase + 16 * CDIM, CDIM);
  v16h b2 = load_fragB(bBase + 32 * CDIM, CDIM);
  v16h b3 = load_fragB(bBase + 48 * CDIM, CDIM);
  for (int k0 = 0; k0 < CDIM; k0 += 32) {
    const int kn = (k0 + 32 == CDIM) ? 0 : (k0 + 32);
    v16h an  = load_fragA(aBase + kn, CDIM);
    v16h bn0 = load_fragB(bBase + kn +  0 * CDIM, CDIM);
    v16h bn1 = load_fragB(bBase + kn + 16 * CDIM, CDIM);
    v16h bn2 = load_fragB(bBase + kn + 32 * CDIM, CDIM);
    v16h bn3 = load_fragB(bBase + kn + 48 * CDIM, CDIM);
    SCHED_FENCE();
    acc[0] = wmma16(a, b0, acc[0]);
    acc[1] = wmma16(a, b1, acc[1]);
    acc[2] = wmma16(a, b2, acc[2]);
    acc[3] = wmma16(a, b3, acc[3]);
    a = an; b0 = bn0; b1 = bn1; b2 = bn2; b3 = bn3;
  }

  const int c = lane & 15, g = lane >> 4;
  #pragma unroll
  for (int r = 0; r < 8; ++r) {
    const int tok = m0 + r + 8 * g;      // always < 9232 (exact tiling)
    #pragma unroll
    for (int t = 0; t < 4; ++t) {
      const int o = n0 + t * 16 + c;
      out[(size_t)tok * CDIM + o] = acc[t][r] + pb[o];
    }
  }
}

extern "C" void kernel_launch(void* const* d_in, const int* in_sizes, int n_in,
                              void* d_out, int out_size, void* d_ws, size_t ws_size,
                              hipStream_t stream)
{
  (void)in_sizes; (void)n_in; (void)out_size; (void)ws_size;
  const float* x  = (const float*)d_in[0];   // [16,577,768]
  const float* wq = (const float*)d_in[1];   // [2304,768]
  const float* wp = (const float*)d_in[2];   // [768,768]
  const float* pb = (const float*)d_in[3];   // [768]
  float* out = (float*)d_out;                // [16,577,768] f32

  char* ws = (char*)d_ws;
  size_t off = 0;
  auto carve = [&](size_t elems) -> _Float16* {
    _Float16* p = (_Float16*)(ws + off);
    off += ((elems * sizeof(_Float16)) + 255) & ~(size_t)255;
    return p;
  };
  _Float16* xh  = carve((size_t)NTOK * CDIM);        // 14.2 MB
  _Float16* wqh = carve((size_t)3 * CDIM * CDIM);    //  3.5 MB
  _Float16* wph = carve((size_t)CDIM * CDIM);        //  1.2 MB
  _Float16* q_h = carve((size_t)BH * NPAD * HDIM);   // 14.9 MB
  _Float16* k_h = carve((size_t)BH * NPAD * HDIM);   // 14.9 MB
  _Float16* vT  = carve((size_t)BH * NPAD * HDIM);   // 14.9 MB (V transposed [BH,D,NPAD])
  _Float16* aoh = carve((size_t)NTOK * CDIM);        // 14.2 MB   (~78 MB total)

  k_cvt<<<1024, 256, 0, stream>>>(x,  xh,  (NTOK * CDIM) / 4);
  k_cvt<<<512,  256, 0, stream>>>(wq, wqh, (3 * CDIM * CDIM) / 4);
  k_cvt<<<256,  256, 0, stream>>>(wp, wph, (CDIM * CDIM) / 4);
  k_qkv<<<(NTOK / 16) * (3 * CDIM / 256), 128, 0, stream>>>(xh, wqh, q_h, k_h, vT);
  k_attn<<<BH * ((NSEQ + 15) / 16), 32, 0, stream>>>(q_h, k_h, vT, aoh);
  k_proj<<<(NTOK / 16) * (CDIM / 256), 128, 0, stream>>>(aoh, wph, pb, out);
}
